// Mukara_27882927685792
// MI455X (gfx1250) — matrix operand order repaired and checked by reading, not compiled, not run on hardware
//
#include <hip/hip_runtime.h>

// ---------------------------------------------------------------------------
// Types / helpers for CDNA5 WMMA (wave32, v_wmma_f32_16x16x32_bf16)
// ---------------------------------------------------------------------------
typedef __attribute__((ext_vector_type(16))) __bf16 bf16x16;
typedef __attribute__((ext_vector_type(8)))  __bf16 bf16x8;
typedef __attribute__((ext_vector_type(4)))  __bf16 bf16x4;
typedef __attribute__((ext_vector_type(2)))  __bf16 bf16x2;
typedef __attribute__((ext_vector_type(8)))  float  f32x8;

__device__ __forceinline__ __bf16 f2bf(float f) {
  unsigned u = __builtin_bit_cast(unsigned, f);
  u += 0x7FFFu + ((u >> 16) & 1u);              // round-to-nearest-even
  unsigned short h = (unsigned short)(u >> 16);
  return __builtin_bit_cast(__bf16, h);
}
__device__ __forceinline__ float bf2f(__bf16 b) {
  unsigned short s = __builtin_bit_cast(unsigned short, b);
  unsigned u = ((unsigned)s) << 16;
  return __builtin_bit_cast(float, u);
}
// Packed f32x2 -> bf16x2 (v_cvt_pk_bf16_f32 when available).
__device__ __forceinline__ bf16x2 pk2(float lo, float hi) {
#if __has_builtin(__builtin_amdgcn_cvt_pk_bf16_f32)
  auto r = __builtin_amdgcn_cvt_pk_bf16_f32(lo, hi);
  return __builtin_bit_cast(bf16x2, r);
#else
  bf16x2 r; r[0] = f2bf(lo); r[1] = f2bf(hi); return r;
#endif
}
__device__ __forceinline__ bf16x8 pack8(float4 a, float4 b) {
  bf16x2 p0 = pk2(a.x, a.y), p1 = pk2(a.z, a.w);
  bf16x2 p2 = pk2(b.x, b.y), p3 = pk2(b.z, b.w);
  bf16x4 q0 = __builtin_shufflevector(p0, p1, 0, 1, 2, 3);
  bf16x4 q1 = __builtin_shufflevector(p2, p3, 0, 1, 2, 3);
  return __builtin_shufflevector(q0, q1, 0, 1, 2, 3, 4, 5, 6, 7);
}

#define WMMA_BF16(A, B, C) \
  __builtin_amdgcn_wmma_f32_16x16x32_bf16(false, (A), false, (B), (short)0, (C), false, false)

// Scalar (guarded) A-fragment load from row-major bf16 (only for tiny F).
__device__ __forceinline__ bf16x16 load_a_bf(const __bf16* tile, int ld, int lane,
                                             int kt, int F) {
  int row = lane & 15, khalf = lane >> 4;
  const __bf16* r = tile + (size_t)row * ld;
  bf16x16 a;
#pragma unroll
  for (int v = 0; v < 8; ++v) {
    int k = kt * 32 + ((v < 4) ? 0 : 16) + khalf * 8 + 2 * (v & 3);
    a[2 * v]     = (k < F)     ? r[k]     : f2bf(0.f);
    a[2 * v + 1] = (k + 1 < F) ? r[k + 1] : f2bf(0.f);
  }
  return a;
}

// Element offset of (row, c) in A-fragment-swizzled activation buffer with
// KTc K-tiles per 16-row tile.  For c aligned to 8, the next 8 columns are the
// next 8 contiguous elements (16-byte packed chunk).
__device__ __forceinline__ size_t afrag_chunk(int row, int c, int KTc) {
  int t16 = row >> 4, rl = row & 15;
  int kt = c >> 5, c5 = c & 31;
  int l = rl + (((c5 >> 3) & 1) << 4);
  int j = (c5 >> 4) << 3;
  return ((size_t)(t16 * KTc + kt) * 32 + l) * 16 + j;
}

// ---------------------------------------------------------------------------
// Small elementwise / gather kernels (bf16 out: k_cvt_bf16 row-major, the others fragment-swizzled)
// ---------------------------------------------------------------------------
__global__ void k_cvt_bf16(const float* __restrict__ in, __bf16* __restrict__ out,
                           long n) {
  long i = (long)blockIdx.x * blockDim.x + threadIdx.x;
  long st = (long)gridDim.x * blockDim.x;
  for (; i < n; i += st) out[i] = f2bf(in[i]);
}

// B_h = wq[h] @ wk[h]^T -> Bcat written directly in B-fragment layout:
// logical Bcat[k=dp][col=h*64+d], tiles (kt=dp>>5, nt=col>>4).
__global__ void k_bcat(const float* __restrict__ wq, const float* __restrict__ wk,
                       __bf16* __restrict__ Bcat) {
  int t = blockIdx.x * blockDim.x + threadIdx.x;
  if (t >= 4 * 64 * 64) return;
  int h = t >> 12, dp = (t >> 6) & 63, d = t & 63;
  const float* q = wq + (size_t)h * 4096 + (size_t)dp * 64;
  const float* k = wk + (size_t)h * 4096 + (size_t)d * 64;
  float s = 0.f;
  for (int e = 0; e < 64; ++e) s += q[e] * k[e];
  int col = h * 64 + d;
  int kt = dp >> 5, lh = (dp >> 3) & 1, j = (((dp >> 4) & 1) << 3) + (dp & 7);
  int nt = col >> 4, l = (col & 15) + 16 * lh;
  Bcat[((size_t)(kt * 16 + nt) * 32 + l) * 16 + j] = f2bf(s);
}

// Xe[e][0:64]=node_emb[n2e[e]], Xe[e][64:128]=edge_emb[e]; fragment layout KT=4
__global__ void k_gather_xe(const float* __restrict__ nemb, const float* __restrict__ eemb,
                            const int* __restrict__ n2e, __bf16* __restrict__ Xe, int E) {
  long total = (long)E * 16;  // 8-column chunks
  long i = (long)blockIdx.x * blockDim.x + threadIdx.x;
  long st = (long)gridDim.x * blockDim.x;
  for (; i < total; i += st) {
    int e = (int)(i >> 4), c = (int)(i & 15) * 8;
    const float* src = (c < 64) ? (nemb + (size_t)n2e[e] * 64 + c)
                                : (eemb + (size_t)e * 64 + (c - 64));
    float4 s0 = ((const float4*)src)[0];
    float4 s1 = ((const float4*)src)[1];
    *(bf16x8*)(Xe + afrag_chunk(e, c, 4)) = pack8(s0, s1);
  }
}

// Xf[e][0:64]=node_emb[u]+node_emb[v], Xf[e][64:128]=edge_emb0; KT=4
__global__ void k_gather_xf(const float* __restrict__ nemb, const __bf16* __restrict__ eemb0,
                            const int* __restrict__ eu, const int* __restrict__ ev,
                            __bf16* __restrict__ Xf, int E) {
  long total = (long)E * 16;
  long i = (long)blockIdx.x * blockDim.x + threadIdx.x;
  long st = (long)gridDim.x * blockDim.x;
  for (; i < total; i += st) {
    int e = (int)(i >> 4), c = (int)(i & 15) * 8;
    bf16x8 o;
    if (c < 64) {
      const float* su = nemb + (size_t)eu[e] * 64 + c;
      const float* sv = nemb + (size_t)ev[e] * 64 + c;
      float4 u0 = ((const float4*)su)[0], u1 = ((const float4*)su)[1];
      float4 v0 = ((const float4*)sv)[0], v1 = ((const float4*)sv)[1];
      float4 a0 = {u0.x + v0.x, u0.y + v0.y, u0.z + v0.z, u0.w + v0.w};
      float4 a1 = {u1.x + v1.x, u1.y + v1.y, u1.z + v1.z, u1.w + v1.w};
      o = pack8(a0, a1);
    } else {
      o = *(const bf16x8*)(eemb0 + (size_t)e * 64 + (c - 64));
    }
    *(bf16x8*)(Xf + afrag_chunk(e, c, 4)) = o;
  }
}

// ---------------------------------------------------------------------------
// Generic WMMA MLP: out = relu(X@W1+b1)@W2+b2, optional residual+LayerNorm.
// Weights staged to LDS pre-swizzled into fragment order (32B/lane loads,
// software-pipelined one fragment ahead of each WMMA).
// ---------------------------------------------------------------------------
template <int KT, int NT2, bool LN>
__global__ void k_mlp(const __bf16* __restrict__ X, int ldx, int F,
                      const float* __restrict__ W1, const float* __restrict__ b1,
                      const float* __restrict__ W2, const float* __restrict__ b2,
                      int DO, const float* __restrict__ res,
                      const float* __restrict__ gamma, const float* __restrict__ beta,
                      float* __restrict__ out, int ldo, int rows) {
  extern __shared__ __align__(32) char smem[];
  constexpr int Kp  = KT * 32;
  constexpr int DOp = NT2 * 16;
  __bf16* w1s = (__bf16*)smem;                 // KT*8 fragment tiles (Kp*128)
  __bf16* w2s = w1s + (size_t)Kp * 128;        // 4*NT2 fragment tiles (128*DOp)
  float*  b1s = (float*)(w2s + 128 * DOp);     // 128
  float*  b2s = b1s + 128;                     // DOp
  float*  gs  = b2s + DOp;                     // 64
  float*  bs  = gs + 64;                       // 64
  __bf16* pw  = (__bf16*)(bs + 64);            // per wave: 16*128 bf16 + 64 f32

  int tid = threadIdx.x, nthr = blockDim.x;
  for (int i = tid; i < Kp * 128; i += nthr) {
    int j = i & 15, l = (i >> 4) & 31, t = i >> 9;
    int kt = t >> 3, nt = t & 7;
    int k = kt * 32 + (j >> 3) * 16 + (l >> 4) * 8 + (j & 7);
    int col = nt * 16 + (l & 15);
    w1s[i] = f2bf((k < F) ? W1[(size_t)k * 128 + col] : 0.f);
  }
  for (int i = tid; i < 128 * DOp; i += nthr) {
    int j = i & 15, l = (i >> 4) & 31, t = i >> 9;
    int kt = t / NT2, nt = t - kt * NT2;
    int k = kt * 32 + (j >> 3) * 16 + (l >> 4) * 8 + (j & 7);
    int col = nt * 16 + (l & 15);
    w2s[i] = f2bf((col < DO) ? W2[(size_t)k * DO + col] : 0.f);
  }
  for (int i = tid; i < 128; i += nthr) b1s[i] = b1[i];
  for (int i = tid; i < DOp; i += nthr) b2s[i] = (i < DO) ? b2[i] : 0.f;
  if constexpr (LN) {
    for (int i = tid; i < 64; i += nthr) { gs[i] = gamma[i]; bs[i] = beta[i]; }
  }
  __syncthreads();

  int lane = tid & 31, wave = tid >> 5;
  int tile = (blockIdx.x * (nthr >> 5) + wave) * 16;
  if (tile >= rows) return;

  __bf16* hstf = pw + (size_t)wave * (16 * 128 + 128);  // hidden, fragment layout
  float*  red  = (float*)(hstf + 16 * 128);             // 64-float reduction area
  int colb = lane & 15, khalf = lane >> 4;

  // ---- GEMM1: hidden = relu(X @ W1 + b1) ----
  bf16x16 afr[KT];
  if constexpr (KT == 1) {
    afr[0] = load_a_bf(X + (size_t)tile * ldx, ldx, lane, 0, F);
  } else {
#pragma unroll
    for (int kt = 0; kt < KT; ++kt)
      afr[kt] = *(const bf16x16*)(X + ((size_t)((tile >> 4) * KT + kt) * 32 + lane) * 16);
  }

  auto ldB1 = [&](int kt, int nt) {
    return *(const bf16x16*)(w1s + ((size_t)(kt * 8 + nt) * 32 + lane) * 16);
  };
  bf16x16 bcur = ldB1(0, 0);
#pragma unroll
  for (int nt = 0; nt < 8; ++nt) {
    f32x8 acc = {0.f, 0.f, 0.f, 0.f, 0.f, 0.f, 0.f, 0.f};
#pragma unroll
    for (int kt = 0; kt < KT; ++kt) {
      bf16x16 bnxt = bcur;
      if (!(nt == 7 && kt == KT - 1))
        bnxt = (kt == KT - 1) ? ldB1(0, nt + 1) : ldB1(kt + 1, nt);
      acc = WMMA_BF16(afr[kt], bcur, acc);
      bcur = bnxt;
    }
    int col = nt * 16 + colb;
    int kt2 = col >> 5, c5 = col & 31;
    int lh = (c5 >> 3) & 1;
    int j  = ((c5 >> 4) << 3) + (((c5 >> 1) & 3) << 1) + (c5 & 1);
    float bb = b1s[col];
#pragma unroll
    for (int v = 0; v < 8; ++v) {
      float h = acc[v] + bb;
      h = h > 0.f ? h : 0.f;
      hstf[(size_t)kt2 * 512 + (v + 8 * khalf + 16 * lh) * 16 + j] = f2bf(h);
    }
  }

  // ---- GEMM2: u = hidden @ W2 + b2 ----
  bf16x16 a2[4];
#pragma unroll
  for (int kt = 0; kt < 4; ++kt)
    a2[kt] = *(const bf16x16*)(hstf + (size_t)kt * 512 + lane * 16);

  auto ldB2 = [&](int kt, int nt) {
    return *(const bf16x16*)(w2s + ((size_t)(kt * NT2 + nt) * 32 + lane) * 16);
  };
  f32x8 acc2[NT2];
  bf16x16 b2cur = ldB2(0, 0);
#pragma unroll
  for (int nt = 0; nt < NT2; ++nt) {
    f32x8 acc = {0.f, 0.f, 0.f, 0.f, 0.f, 0.f, 0.f, 0.f};
#pragma unroll
    for (int kt = 0; kt < 4; ++kt) {
      bf16x16 bnxt = b2cur;
      if (!(nt == NT2 - 1 && kt == 3))
        bnxt = (kt == 3) ? ldB2(0, nt + 1) : ldB2(kt + 1, nt);
      acc = WMMA_BF16(a2[kt], b2cur, acc);
      b2cur = bnxt;
    }
    acc2[nt] = acc;
  }

  if constexpr (!LN) {
#pragma unroll
    for (int nt = 0; nt < NT2; ++nt) {
      int col = nt * 16 + colb;
      if (col < DO) {
        float bb = b2s[col];
#pragma unroll
        for (int v = 0; v < 8; ++v)
          out[(size_t)(tile + v + 8 * khalf) * ldo + col] = acc2[nt][v] + bb;
      }
    }
  } else {
    // residual + LayerNorm over 64 features (DO==64, NT2==4)
    float* xln = (float*)hstf;  // reuse: 16x64 f32
#pragma unroll
    for (int nt = 0; nt < 4; ++nt) {
      int col = nt * 16 + colb;
      float bb = b2s[col];
#pragma unroll
      for (int v = 0; v < 8; ++v) {
        int M = v + 8 * khalf;
        xln[M * 64 + col] = acc2[nt][v] + bb + res[(size_t)(tile + M) * 64 + col];
      }
    }
    int rrow = lane & 15;
    float s = 0.f, s2 = 0.f;
    const float* xr = xln + rrow * 64 + khalf * 32;
#pragma unroll
    for (int c = 0; c < 32; ++c) { float x = xr[c]; s += x; s2 += x * x; }
    red[lane] = s; red[32 + lane] = s2;
    float st  = s  + red[lane ^ 16];
    float s2t = s2 + red[32 + (lane ^ 16)];
    float mean = st * (1.f / 64.f);
    float var  = s2t * (1.f / 64.f) - mean * mean;
    float rstd = rsqrtf(var + 1e-6f);
    if (lane < 16) { red[2 * rrow] = mean; red[2 * rrow + 1] = rstd; }
#pragma unroll
    for (int i = 0; i < 32; ++i) {
      int idx = lane + 32 * i;
      int row = idx >> 6, col = idx & 63;
      float m = red[2 * row], rs = red[2 * row + 1];
      float val = (xln[row * 64 + col] - m) * rs * gs[col] + bs[col];
      out[(size_t)(tile + row) * ldo + col] = val;
    }
  }
}

// ---------------------------------------------------------------------------
// Edge->node attention (s_k = ke_k . (x @ Bcat_h), Bcat precomputed).
// 4 waves/block, 16 nodes/wave.  Bcat B-fragments read from global
// (L2-resident, 32KB), software-pipelined.  Xn written fragment-swizzled.
// ---------------------------------------------------------------------------
__global__ void k_attn(const float* __restrict__ nemb, const float* __restrict__ eemb,
                       const int* __restrict__ inc_edges,
                       const unsigned char* __restrict__ inc_mask,
                       const __bf16* __restrict__ BcatG,
                       __bf16* __restrict__ Xn, int N) {
  extern __shared__ __align__(32) char smem[];
  int tid = threadIdx.x, nthr = blockDim.x;
  int lane = tid & 31, wave = tid >> 5;
  int tile = (blockIdx.x * (nthr >> 5) + wave) * 16;
  if (tile >= N) return;

  const size_t pwBytes = 16 * 8 * 64 * 2 + 16 * 256 * 4 + 512;
  char* base = smem + (size_t)wave * pwBytes;
  __bf16* ke  = (__bf16*)base;                     // [node][k][d] bf16 (16KB)
  float*  Vs  = (float*)(base + 16 * 8 * 64 * 2);  // [node][256] f32 (16KB)
  int*    idxs = (int*)(base + 16 * 8 * 64 * 2 + 16 * 256 * 4);  // [node][k]
  int colb = lane & 15, khalf = lane >> 4;

  for (int i = lane; i < 128; i += 32) idxs[i] = inc_edges[(size_t)tile * 8 + i];

  // A fragments of x (node_emb rows, f32 -> bf16), vectorized loads
  bf16x16 a0, a1;
  {
    const float* r = nemb + (size_t)(tile + (lane & 15)) * 64;
#pragma unroll
    for (int kt = 0; kt < 2; ++kt) {
      const float4* p0 = (const float4*)(r + kt * 32 + khalf * 8);
      const float4* p1 = (const float4*)(r + kt * 32 + khalf * 8 + 16);
      float4 x0 = p0[0], x1 = p0[1], y0 = p1[0], y1 = p1[1];
      bf16x8 lo = pack8(x0, x1), hi = pack8(y0, y1);
      bf16x16 a = __builtin_shufflevector(lo, hi, 0, 1, 2, 3, 4, 5, 6, 7,
                                          8, 9, 10, 11, 12, 13, 14, 15);
      if (kt == 0) a0 = a; else a1 = a;
    }
  }

  // Vcat = x @ Bcat : [16,64] x [64,256], V kept f32 in LDS (pipelined B)
  auto ldB = [&](int kt, int nt) {
    return *(const bf16x16*)(BcatG + ((size_t)(kt * 16 + nt) * 32 + lane) * 16);
  };
  bf16x16 b0 = ldB(0, 0), b1v = ldB(1, 0);
#pragma unroll
  for (int nt = 0; nt < 16; ++nt) {
    bf16x16 n0 = b0, n1 = b1v;
    if (nt < 15) { n0 = ldB(0, nt + 1); n1 = ldB(1, nt + 1); }
    f32x8 acc = {0.f, 0.f, 0.f, 0.f, 0.f, 0.f, 0.f, 0.f};
    acc = WMMA_BF16(a0, b0, acc);
    acc = WMMA_BF16(a1, b1v, acc);
    int col = nt * 16 + colb;
#pragma unroll
    for (int v = 0; v < 8; ++v) Vs[(v + 8 * khalf) * 256 + col] = acc[v];
    b0 = n0; b1v = n1;
  }

  // Gather incident-edge embeddings into LDS (packed 16B stores)
  for (int i = lane; i < 16 * 8 * 8; i += 32) {
    int n = i >> 6, k = (i >> 3) & 7, c8 = (i & 7) * 8;
    const float* src = eemb + (size_t)idxs[n * 8 + k] * 64 + c8;
    float4 s0 = ((const float4*)src)[0];
    float4 s1 = ((const float4*)src)[1];
    *(bf16x8*)(ke + (size_t)(n * 8 + k) * 64 + c8) = pack8(s0, s1);
  }

  // Scores + softmax(K=8) + weighted sum; lane: node=lane&15, 2 heads
  const float scale = 0.125f;  // 1/sqrt(64)
  int node = lane & 15;
  for (int h2 = 0; h2 < 2; ++h2) {
    int h = khalf * 2 + h2;
    float4 vv[16];
    const float4* vr4 = (const float4*)(Vs + node * 256 + h * 64);
#pragma unroll
    for (int q = 0; q < 16; ++q) vv[q] = vr4[q];

    float s[8];
#pragma unroll
    for (int k = 0; k < 8; ++k) {
      const bf16x8* kr = (const bf16x8*)(ke + (size_t)(node * 8 + k) * 64);
      float d = 0.f;
#pragma unroll
      for (int c8 = 0; c8 < 8; ++c8) {
        bf16x8 kv = kr[c8];
        float4 v0 = vv[c8 * 2], v1 = vv[c8 * 2 + 1];
        d += bf2f(kv[0]) * v0.x + bf2f(kv[1]) * v0.y +
             bf2f(kv[2]) * v0.z + bf2f(kv[3]) * v0.w +
             bf2f(kv[4]) * v1.x + bf2f(kv[5]) * v1.y +
             bf2f(kv[6]) * v1.z + bf2f(kv[7]) * v1.w;
      }
      unsigned char m = inc_mask[(size_t)(tile + node) * 8 + k];
      s[k] = m ? d * scale : -1e9f;
    }
    float mx = s[0];
#pragma unroll
    for (int k = 1; k < 8; ++k) mx = fmaxf(mx, s[k]);
    float a[8]; float se = 0.f;
#pragma unroll
    for (int k = 0; k < 8; ++k) { a[k] = __expf(s[k] - mx); se += a[k]; }
    float inv = 1.f / se;
#pragma unroll
    for (int k = 0; k < 8; ++k) a[k] *= inv;

#pragma unroll
    for (int c8 = 0; c8 < 8; ++c8) {
      float o[8] = {0.f, 0.f, 0.f, 0.f, 0.f, 0.f, 0.f, 0.f};
#pragma unroll
      for (int k = 0; k < 8; ++k) {
        bf16x8 kv = *(const bf16x8*)(ke + (size_t)(node * 8 + k) * 64 + c8 * 8);
        float ak = a[k];
#pragma unroll
        for (int t = 0; t < 8; ++t) o[t] += ak * bf2f(kv[t]);
      }
      float4 o0 = {o[0], o[1], o[2], o[3]};
      float4 o1 = {o[4], o[5], o[6], o[7]};
      *(bf16x8*)(Xn + afrag_chunk(tile + node, h * 64 + c8 * 8, 10)) = pack8(o0, o1);
    }
  }

  // append node_emb (cols 256..319), swizzled chunk stores
  for (int i = lane; i < 128; i += 32) {
    int n = i >> 3, c8 = (i & 7) * 8;
    const float* src = nemb + (size_t)(tile + n) * 64 + c8;
    float4 s0 = ((const float4*)src)[0];
    float4 s1 = ((const float4*)src)[1];
    *(bf16x8*)(Xn + afrag_chunk(tile + n, 256 + c8, 10)) = pack8(s0, s1);
  }
}

// ---------------------------------------------------------------------------
// Host side
// ---------------------------------------------------------------------------
static inline int cdiv_i(long a, long b) { return (int)((a + b - 1) / b); }

template <int KT, int NT2, bool LN>
static void launch_mlp(const __bf16* X, int ldx, int F, const float* W1, const float* b1,
                       const float* W2, const float* b2, int DO, const float* res,
                       const float* gamma, const float* beta, float* out, int ldo,
                       int rows, hipStream_t stream) {
  size_t smem = (size_t)KT * 32 * 128 * 2 + (size_t)128 * NT2 * 16 * 2 +
                128 * 4 + NT2 * 16 * 4 + 64 * 8 + 8 * (16 * 128 * 2 + 64 * 4);
  dim3 grid(cdiv_i(rows, 128)), block(256);
  k_mlp<KT, NT2, LN><<<grid, block, smem, stream>>>(X, ldx, F, W1, b1, W2, b2, DO,
                                                    res, gamma, beta, out, ldo, rows);
}

extern "C" void kernel_launch(void* const* d_in, const int* in_sizes, int n_in,
                              void* d_out, int out_size, void* d_ws, size_t ws_size,
                              hipStream_t stream) {
  const float* node_feats = (const float*)d_in[0];
  const float* edge_feats = (const float*)d_in[1];
  const int* n2e_node = (const int*)d_in[2];
  const int* inc_edges = (const int*)d_in[3];
  const unsigned char* inc_mask = (const unsigned char*)d_in[4];
  const int* edge_u = (const int*)d_in[5];
  const int* edge_v = (const int*)d_in[6];
  const float* ne_w1 = (const float*)d_in[7];
  const float* ne_b1 = (const float*)d_in[8];
  const float* ne_w2 = (const float*)d_in[9];
  const float* ne_b2 = (const float*)d_in[10];
  const float* ee_w1 = (const float*)d_in[11];
  const float* ee_b1 = (const float*)d_in[12];
  const float* ee_w2 = (const float*)d_in[13];
  const float* ee_b2 = (const float*)d_in[14];
  const float* n2e_w1 = (const float*)d_in[15];
  const float* n2e_b1 = (const float*)d_in[16];
  const float* n2e_w2 = (const float*)d_in[17];
  const float* n2e_b2 = (const float*)d_in[18];
  const float* wq = (const float*)d_in[19];
  const float* wk = (const float*)d_in[20];
  const float* e2n_w1 = (const float*)d_in[21];
  const float* e2n_b1 = (const float*)d_in[22];
  const float* e2n_w2 = (const float*)d_in[23];
  const float* e2n_b2 = (const float*)d_in[24];
  const float* en_gamma = (const float*)d_in[25];
  const float* en_beta = (const float*)d_in[26];
  const float* nn_gamma = (const float*)d_in[27];
  const float* nn_beta = (const float*)d_in[28];
  const float* out_w1 = (const float*)d_in[29];
  const float* out_b1 = (const float*)d_in[30];
  const float* out_w2 = (const float*)d_in[31];
  const float* out_b2 = (const float*)d_in[32];

  const int FN = 10, FE = 17;
  const int N = in_sizes[0] / FN;
  const int E = in_sizes[1] / FE;

  char* ws = (char*)d_ws;
  size_t off = 0;
  auto alloc = [&](size_t bytes) -> void* {
    void* p = ws + off;
    off += (bytes + 255) & ~(size_t)255;
    return p;
  };
  float* node_emb = (float*)alloc((size_t)N * 64 * 4);
  float* edge_emb = (float*)alloc((size_t)E * 64 * 4);
  __bf16* eemb0   = (__bf16*)alloc((size_t)E * 64 * 2);
  __bf16* Bcat    = (__bf16*)alloc((size_t)64 * 256 * 2);
  size_t bigElems = (size_t)E * 128;
  size_t xnElems  = (size_t)N * 320;
  if (xnElems > bigElems) bigElems = xnElems;
  __bf16* Xbig = (__bf16*)alloc(bigElems * 2);
  __bf16* Xnf  = Xbig;                       // overlap (dead before layer loop)
  __bf16* Xef  = Xbig + (size_t)N * FN;

  dim3 blk(256);
  // 1) feature conversion to bf16 (row-major; embedders use scalar A path)
  k_cvt_bf16<<<dim3(cdiv_i((long)N * FN, 256)), blk, 0, stream>>>(node_feats, Xnf, (long)N * FN);
  k_cvt_bf16<<<dim3(cdiv_i((long)E * FE, 256)), blk, 0, stream>>>(edge_feats, Xef, (long)E * FE);
  // 2) node / edge embedders
  launch_mlp<1, 4, false>(Xnf, FN, FN, ne_w1, ne_b1, ne_w2, ne_b2, 64,
                          nullptr, nullptr, nullptr, node_emb, 64, N, stream);
  launch_mlp<1, 4, false>(Xef, FE, FE, ee_w1, ee_b1, ee_w2, ee_b2, 64,
                          nullptr, nullptr, nullptr, edge_emb, 64, E, stream);
  // 3) save initial edge embedding ("center" of the output MLP)
  k_cvt_bf16<<<dim3(cdiv_i((long)E * 64, 256)), blk, 0, stream>>>(edge_emb, eemb0, (long)E * 64);
  // 4) fused attention matrices B_h = wq_h @ wk_h^T (B-fragment layout)
  k_bcat<<<dim3(64), blk, 0, stream>>>(wq, wk, Bcat);

  const size_t attn_smem = 4 * ((size_t)16 * 8 * 64 * 2 + 16 * 256 * 4 + 512);

  for (int l = 0; l < 3; ++l) {
    // node2edge: Xe = [node_emb[src] | edge_emb]; MLP + residual + LN
    k_gather_xe<<<dim3(cdiv_i((long)E * 16, 256)), blk, 0, stream>>>(
        node_emb, edge_emb, n2e_node, Xbig, E);
    launch_mlp<4, 4, true>(Xbig, 128, 128, n2e_w1, n2e_b1, n2e_w2, n2e_b2, 64,
                           edge_emb, en_gamma, en_beta, edge_emb, 64, E, stream);
    // edge2node: attention heads -> Xn[N][320] (fragment layout)
    k_attn<<<dim3(cdiv_i(N, 64)), dim3(128), attn_smem, stream>>>(
        node_emb, edge_emb, inc_edges, inc_mask, Bcat, Xbig, N);
    launch_mlp<10, 4, true>(Xbig, 320, 320, e2n_w1, e2n_b1, e2n_w2, e2n_b2, 64,
                            node_emb, nn_gamma, nn_beta, node_emb, 64, N, stream);
  }

  // final head: Xf = [node_emb[u]+node_emb[v] | edge_emb0]; MLP 128->128->1
  k_gather_xf<<<dim3(cdiv_i((long)E * 16, 256)), blk, 0, stream>>>(
      node_emb, eemb0, edge_u, edge_v, Xbig, E);
  launch_mlp<4, 1, false>(Xbig, 128, 128, out_w1, out_b1, out_w2, out_b2, 1,
                          nullptr, nullptr, nullptr, (float*)d_out, 1, E, stream);
  (void)n_in; (void)out_size; (void)ws_size;
}
